// FMCBOWModel_11871289606266
// MI455X (gfx1250) — matrix operand order repaired and checked by reading, not compiled, not run on hardware
//
#include <hip/hip_runtime.h>
#include <hip/hip_bf16.h>

#define EMB_DIM 128
#define CTX     8
#define NEG     5
#define BATCH   16384

typedef __attribute__((ext_vector_type(16))) _Float16 v16h;
typedef __attribute__((ext_vector_type(8)))  float    v8f;

__device__ __forceinline__ float log_sigmoid(float x) {
    // log(sigmoid(x)) = min(x,0) - log1p(exp(-|x|))  (stable)
    return fminf(x, 0.0f) - __logf(1.0f + __expf(-fabsf(x)));
}

__global__ void zero_out_kernel(float* out) {
    if (threadIdx.x == 0 && blockIdx.x == 0) out[0] = 0.0f;
}

__global__ void __launch_bounds__(256)
fmcbow_fused(const int*   __restrict__ pos_u,
             const int*   __restrict__ pos_w,
             const int*   __restrict__ neg_w,
             const float* __restrict__ u_emb,
             const float* __restrict__ w_emb,
             const float* __restrict__ vmat,
             float*       __restrict__ out)
{
    const int tid  = blockIdx.x * blockDim.x + threadIdx.x;
    const int pair = tid >> 5;                 // one sample pair per wave32
    const int lane = threadIdx.x & 31;
    if (pair * 2 >= BATCH) return;

    const int m  = lane & 15;                  // A-row this lane feeds / B,C,D column it owns
    const int hi = lane >> 4;                  // K-half selector within a 32-K chunk

    // ---------------- Phase 1: build A (ctx) and B (V^T) f16 tiles ----------------
    // A row m: rows 0-7 = sample 2p (c = m), rows 8-15 = sample 2p+1 (c = m-8)
    const int   sA   = pair * 2 + (m >> 3);
    const int   cA   = m & 7;
    const float* urow = u_emb + (size_t)pos_u[sA * CTX + cA] * EMB_DIM;

    v16h a[4], a2[4];
    #pragma unroll
    for (int q = 0; q < 4; ++q) {
        const int b0 = q * 32 + hi * 8;        // ISA A-layout: e<8 -> K=b0+e, e>=8 -> K=b0+8+e
        #pragma unroll
        for (int e = 0; e < 8; ++e) {
            float x0 = urow[b0 + e];
            float x1 = urow[b0 + 16 + e];
            a[q][e]     = (_Float16)x0;
            a[q][8 + e] = (_Float16)x1;
            // ctx^2 ~ 1.5e-5 is f16-subnormal; scale by 16^2 and descale accumulator later
            float s0 = x0 * 16.0f, s1 = x1 * 16.0f;
            a2[q][e]     = (_Float16)(s0 * s0);
            a2[q][8 + e] = (_Float16)(s1 * s1);
        }
    }

    // B = V^T (128x16): lane owns column n=m; element e -> K = q*32 + hi*8 + e + (e & 8)
    const float* vrow = vmat + m * EMB_DIM;
    v16h b[4], b2[4];
    #pragma unroll
    for (int q = 0; q < 4; ++q) {
        const int k0 = q * 32 + hi * 8;
        #pragma unroll
        for (int e = 0; e < 16; ++e) {
            float x = vrow[k0 + e + (e & 8)];
            b[q][e]  = (_Float16)x;
            b2[q][e] = (_Float16)(x * x);
        }
    }

    // ---------------- WMMA: ip1 = ctx @ V^T ; ip2 = ctx^2 @ (V^2)^T ----------------
    v8f acc1 = {0.f, 0.f, 0.f, 0.f, 0.f, 0.f, 0.f, 0.f};
    v8f acc2 = {0.f, 0.f, 0.f, 0.f, 0.f, 0.f, 0.f, 0.f};
    #pragma unroll
    for (int q = 0; q < 4; ++q) {
        acc1 = __builtin_amdgcn_wmma_f32_16x16x32_f16(false, a[q],  false, b[q],
                                                      (short)0, acc1, false, false);
        acc2 = __builtin_amdgcn_wmma_f32_16x16x32_f16(false, a2[q], false, b2[q],
                                                      (short)0, acc2, false, false);
    }

    // fm = 0.5 * sum_{c,v} (ip1^2 - ip2). Lanes 0-15 hold the even sample's 8x16
    // tile (VGPR r = row r), lanes 16-31 the odd sample's.
    float t = 0.0f;
    #pragma unroll
    for (int r = 0; r < 8; ++r)
        t += acc1[r] * acc1[r] - acc2[r] * (1.0f / 256.0f);
    #pragma unroll
    for (int mm = 1; mm < 16; mm <<= 1)
        t += __shfl_xor(t, mm, 32);
    const float fm = 0.5f * t;   // lanes 0-15: fm(2p), lanes 16-31: fm(2p+1)

    // ---------------- Phase 2: ctx sum + dot products (coalesced, L2-hot) ----------------
    const int s     = lane >> 4;               // matches which fm this half-wave holds
    const int smp   = pair * 2 + s;
    const int dbase = (lane & 15) * 8;         // 8 consecutive dims per lane -> 512B/row coalesced

    float ctxsum[8] = {0.f, 0.f, 0.f, 0.f, 0.f, 0.f, 0.f, 0.f};
    #pragma unroll
    for (int c = 0; c < CTX; ++c) {
        const float* r = u_emb + (size_t)pos_u[smp * CTX + c] * EMB_DIM + dbase;
        #pragma unroll
        for (int e = 0; e < 8; ++e) ctxsum[e] += r[e];
    }

    float pwsum = 0.f, dotp = 0.f;
    {
        const float* pw = w_emb + (size_t)pos_w[smp] * EMB_DIM + dbase;
        #pragma unroll
        for (int e = 0; e < 8; ++e) {
            float x = pw[e];
            pwsum += x;
            dotp = fmaf(x, ctxsum[e], dotp);
        }
    }

    float nwsum = 0.f, dotn = 0.f;
    #pragma unroll
    for (int k = 0; k < NEG; ++k) {
        const float* nw = w_emb + (size_t)neg_w[smp * NEG + k] * EMB_DIM + dbase;
        #pragma unroll
        for (int e = 0; e < 8; ++e) {
            float x = nw[e];
            nwsum += x;
            dotn = fmaf(x, ctxsum[e], dotn);
        }
    }

    #pragma unroll
    for (int mm = 1; mm < 16; mm <<= 1) {
        pwsum += __shfl_xor(pwsum, mm, 32);
        dotp  += __shfl_xor(dotp,  mm, 32);
        nwsum += __shfl_xor(nwsum, mm, 32);
        dotn  += __shfl_xor(dotn,  mm, 32);
    }

    // dot(u, w) = dot(ctxsum, w) + (CTX*fm) * sum(w)   since u = ctxsum + CTX*fm
    const float nC     = (float)CTX;
    const float posdot = dotp + nC * fm * pwsum;
    const float negdot = dotn + nC * fm * nwsum;
    const float contrib = log_sigmoid(posdot) + log_sigmoid(-negdot);

    if ((lane & 15) == 0)
        atomicAdd(out, -contrib);              // loss = -(sum of scores)
}

extern "C" void kernel_launch(void* const* d_in, const int* in_sizes, int n_in,
                              void* d_out, int out_size, void* d_ws, size_t ws_size,
                              hipStream_t stream) {
    (void)in_sizes; (void)n_in; (void)out_size; (void)d_ws; (void)ws_size;

    const int*   pos_u = (const int*)d_in[0];
    const int*   pos_w = (const int*)d_in[1];
    const int*   neg_w = (const int*)d_in[2];
    const float* u_emb = (const float*)d_in[3];
    const float* w_emb = (const float*)d_in[4];
    const float* vmat  = (const float*)d_in[5];
    float*       out   = (float*)d_out;

    zero_out_kernel<<<1, 1, 0, stream>>>(out);

    const int pairs = BATCH / 2;               // 8192 waves, one pair each
    const int threads = 256;                   // 8 waves per block
    const int blocks = (pairs * 32) / threads; // 1024
    fmcbow_fused<<<blocks, threads, 0, stream>>>(pos_u, pos_w, neg_w,
                                                 u_emb, w_emb, vmat, out);
}
